// CrossAttention_14594298872172
// MI455X (gfx1250) — compile-verified, matrix operand order checked by reading, not run on hardware
//
#include <hip/hip_runtime.h>
#include <hip/hip_bf16.h>

typedef __attribute__((ext_vector_type(16))) __bf16 v16bf;
typedef __attribute__((ext_vector_type(8)))  __bf16 v8bf;
typedef __attribute__((ext_vector_type(8)))  float  v8f;

typedef unsigned int u32x4 __attribute__((ext_vector_type(4)));
typedef int          i32x8 __attribute__((ext_vector_type(8)));
typedef int          i32x4 __attribute__((ext_vector_type(4)));

#define EMB   768
#define HEADS 8
#define HDIM  96
#define BATCH 4
#define SEQ   2048
#define MROWS (BATCH * SEQ)   /* 8192 */

// ---------------------------------------------------------------------------
// Load a 16x32 bf16 WMMA fragment (A-layout; B uses the symmetric pattern).
// lane L: row = row0 + (L&15); 8 halves at k0+(L>>4)*8, 8 more at +16.
// ---------------------------------------------------------------------------
__device__ __forceinline__ v16bf load_frag_b128(const __bf16* __restrict__ p,
                                                int ld, int row0, int k0) {
  const int lane = threadIdx.x & 31;
  const __bf16* r = p + (size_t)(row0 + (lane & 15)) * (size_t)ld
                      + (size_t)(k0 + ((lane >> 4) << 3));
  v8bf lo = *reinterpret_cast<const v8bf*>(r);
  v8bf hi = *reinterpret_cast<const v8bf*>(r + 16);
  v16bf f;
#pragma unroll
  for (int i = 0; i < 8; ++i) { f[i] = lo[i]; f[i + 8] = hi[i]; }
  return f;
}

// Same fragment pattern but sourced from a [64][32] LDS tile (ds_load_b128).
__device__ __forceinline__ v16bf lds_frag_b128(const __bf16 (*ldsW)[32], int row0) {
  const int lane = threadIdx.x & 31;
  const __bf16* r = &ldsW[row0 + (lane & 15)][(lane >> 4) << 3];
  v8bf lo = *reinterpret_cast<const v8bf*>(r);
  v8bf hi = *reinterpret_cast<const v8bf*>(r + 16);
  v16bf f;
#pragma unroll
  for (int i = 0; i < 8; ++i) { f[i] = lo[i]; f[i + 8] = hi[i]; }
  return f;
}

// ---------------------------------------------------------------------------
// fp32 -> bf16 down-convert
// ---------------------------------------------------------------------------
__global__ void cvt_f32_bf16(const float* __restrict__ src,
                             __bf16* __restrict__ dst, int n) {
  int i = blockIdx.x * blockDim.x + threadIdx.x;
  if (i < n) dst[i] = (__bf16)src[i];
}

// ---------------------------------------------------------------------------
// TDM: stage a 64x32 bf16 weight tile (rows = N, cols = K) into LDS.
// D# per CDNA5 ISA ch.8: group0 {count, lds_addr, global_addr, type=2},
// group1 {data_size=2B, tensor dims 768x768, tile 32x64, dim0_stride=768}.
// Issued by wave 0 only; completion via s_wait_tensorcnt then block barrier.
// 6-arg builtin form (clang-23 / therock-10.0): extra zero-filled i32x8 group.
// ---------------------------------------------------------------------------
__device__ __forceinline__ void stage_w_tile(const __bf16* __restrict__ W,
                                             __bf16 (*ldsW)[32],
                                             int tileN, int k0) {
#if __has_builtin(__builtin_amdgcn_tensor_load_to_lds) && \
    __has_builtin(__builtin_amdgcn_s_wait_tensorcnt)
  if ((threadIdx.x >> 5) == 0) {
    const unsigned long long ga =
        (unsigned long long)(const void*)(W + (size_t)tileN * EMB + k0);
    // generic pointer into LDS: addr[31:0] is the LDS byte offset (aperture rule)
    const unsigned lds_addr = (unsigned)(unsigned long long)(void*)&ldsW[0][0];

    u32x4 g0;
    g0[0] = 1u;                                   // count=1, user descriptor
    g0[1] = lds_addr;                             // lds_addr[31:0]
    g0[2] = (unsigned)ga;                         // global_addr[31:0]
    g0[3] = (unsigned)((ga >> 32) & 0x1FFFFFFull) // global_addr[56:32]
          | (2u << 30);                           // type=2 ("image")

    i32x8 g1;
    g1[0] = 0x10000;           // workgroup_mask=0, data_size=1 (2 bytes)
    g1[1] = (int)(768u << 16); // tensor_dim0[15:0]  (K extent)
    g1[2] = (int)(768u << 16); // tensor_dim0 hi=0 | tensor_dim1[15:0]
    g1[3] = (int)(32u << 16);  // tensor_dim1 hi=0 | tile_dim0=32 (K)
    g1[4] = 64;                // tile_dim1=64 (N rows) | tile_dim2=0
    g1[5] = 768;               // tensor_dim0_stride[31:0]
    g1[6] = 0;                 // stride hi | tensor_dim1_stride lo
    g1[7] = 0;

    i32x4 gz4 = {0, 0, 0, 0};
    i32x8 gz8 = {0, 0, 0, 0, 0, 0, 0, 0};
    __builtin_amdgcn_tensor_load_to_lds(g0, g1, gz4, gz4, gz8, 0);
    __builtin_amdgcn_s_wait_tensorcnt(0);
  }
#else
  // Fallback: cooperative 256-thread copy (64 rows x 4 chunks of 16B)
  const int t = threadIdx.x;
  const int row = t >> 2, ch = (t & 3) << 3;
  *reinterpret_cast<v8bf*>(&ldsW[row][ch]) =
      *reinterpret_cast<const v8bf*>(W + (size_t)(tileN + row) * EMB + k0 + ch);
#endif
  __syncthreads();
}

// ---------------------------------------------------------------------------
// GEMM: C[M=8192, N=768] = X(bf16) * W(bf16,[N,K])^T + bias
// OUT_MODE 0: fp32 flat [M,768]; 1: bf16 [B,H,SEQ,96]; 2: bf16 [B,H,96,SEQ].
// Block = 8 waves; block tile 256(M) x 64(N); wave tile 32x64.
// Weight tile 64x32 staged in LDS via TDM, reused by all 8 waves.
// ---------------------------------------------------------------------------
template <int OUT_MODE>
__global__ __launch_bounds__(256)
void gemm_wmma(const __bf16* __restrict__ X, const __bf16* __restrict__ W,
               const float* __restrict__ bias, void* __restrict__ out) {
  __shared__ __bf16 ldsW[64][32];

  const int lane  = threadIdx.x & 31;
  const int wv    = threadIdx.x >> 5;
  const int tileM = blockIdx.y * 256 + wv * 32;
  const int tileN = blockIdx.x * 64;

  v8f acc[2][4] = {};
  for (int k0 = 0; k0 < EMB; k0 += 32) {
    stage_w_tile(W, ldsW, tileN, k0);

    const v16bf a0 = load_frag_b128(X, EMB, tileM, k0);
    const v16bf a1 = load_frag_b128(X, EMB, tileM + 16, k0);
#pragma unroll
    for (int t = 0; t < 4; ++t) {
      const v16bf bfr = lds_frag_b128(ldsW, 16 * t);
      acc[0][t] = __builtin_amdgcn_wmma_f32_16x16x32_bf16(
          false, a0, false, bfr, (short)0, acc[0][t], false, false);
      acc[1][t] = __builtin_amdgcn_wmma_f32_16x16x32_bf16(
          false, a1, false, bfr, (short)0, acc[1][t], false, false);
    }
    __syncthreads();   // protect LDS tile before next stage overwrites it
  }

#pragma unroll
  for (int i = 0; i < 2; ++i)
#pragma unroll
    for (int t = 0; t < 4; ++t) {
      const int C = tileN + 16 * t + (lane & 15);
      const float bia = bias[C];
#pragma unroll
      for (int r = 0; r < 8; ++r) {
        const int R = tileM + 16 * i + ((lane >> 4) << 3) + r;
        const float v = acc[i][t][r] + bia;
        if (OUT_MODE == 0) {
          ((float*)out)[(size_t)R * EMB + C] = v;
        } else {
          const int b = R / SEQ, n = R % SEQ;
          const int h = C / HDIM, d = C % HDIM;
          if (OUT_MODE == 1)
            ((__bf16*)out)[(((size_t)(b * HEADS + h)) * SEQ + n) * HDIM + d] = (__bf16)v;
          else
            ((__bf16*)out)[(((size_t)(b * HEADS + h)) * HDIM + d) * SEQ + n] = (__bf16)v;
        }
      }
    }
}

// ---------------------------------------------------------------------------
// Flash attention per (b,h): S = QK^T * (1/sqrt(EMB)), online softmax, O = PV.
// Grid: (SEQ/128, B*H). Each wave owns a 16-row Q tile; 32-KV chunks.
// ---------------------------------------------------------------------------
__global__ __launch_bounds__(256)
void attn_wmma(const __bf16* __restrict__ Qb, const __bf16* __restrict__ Kb,
               const __bf16* __restrict__ Vt, __bf16* __restrict__ Ob) {
  __shared__ __bf16 ldsP[8][16][32];

  const int lane = threadIdx.x & 31;
  const int wv   = threadIdx.x >> 5;
  const int bh   = blockIdx.y;               // 0..31
  const int b    = bh / HEADS, h = bh % HEADS;
  const int q0   = blockIdx.x * 128 + wv * 16;

  const __bf16* Qh = Qb + (size_t)bh * SEQ * HDIM;
  const __bf16* Kh = Kb + (size_t)bh * SEQ * HDIM;
  const __bf16* Vh = Vt + (size_t)bh * HDIM * SEQ;   // [96, SEQ]

  v16bf qf[3];
#pragma unroll
  for (int c = 0; c < 3; ++c) qf[c] = load_frag_b128(Qh, HDIM, q0, 32 * c);

  v8f o[6] = {};
  float mstate[8], lstate[8];
#pragma unroll
  for (int r = 0; r < 8; ++r) { mstate[r] = -3.0e38f; lstate[r] = 0.0f; }

  const float scale = 0.036084391824351615f;  // 1/sqrt(768)

  for (int kv0 = 0; kv0 < SEQ; kv0 += 32) {
    v8f s[2] = {};
#pragma unroll
    for (int j = 0; j < 2; ++j)
#pragma unroll
      for (int c = 0; c < 3; ++c) {
        v16bf kf = load_frag_b128(Kh, HDIM, kv0 + 16 * j, 32 * c);
        s[j] = __builtin_amdgcn_wmma_f32_16x16x32_bf16(
            false, qf[c], false, kf, (short)0, s[j], false, false);
      }

    float mnew[8], corr[8];
#pragma unroll
    for (int r = 0; r < 8; ++r) {
      s[0][r] *= scale; s[1][r] *= scale;
      float t = fmaxf(s[0][r], s[1][r]);
      t = fmaxf(t, __shfl_xor(t, 1, 32));
      t = fmaxf(t, __shfl_xor(t, 2, 32));
      t = fmaxf(t, __shfl_xor(t, 4, 32));
      t = fmaxf(t, __shfl_xor(t, 8, 32));
      mnew[r] = fmaxf(mstate[r], t);
      corr[r] = __expf(mstate[r] - mnew[r]);
      mstate[r] = mnew[r];
    }
#pragma unroll
    for (int r = 0; r < 8; ++r) {
      const float p0 = __expf(s[0][r] - mnew[r]);
      const float p1 = __expf(s[1][r] - mnew[r]);
      s[0][r] = p0; s[1][r] = p1;
      float rs = p0 + p1;
      rs += __shfl_xor(rs, 1, 32);
      rs += __shfl_xor(rs, 2, 32);
      rs += __shfl_xor(rs, 4, 32);
      rs += __shfl_xor(rs, 8, 32);
      lstate[r] = lstate[r] * corr[r] + rs;
    }
#pragma unroll
    for (int co = 0; co < 6; ++co)
#pragma unroll
      for (int r = 0; r < 8; ++r) o[co][r] *= corr[r];

    // transpose P (C-layout) -> A-fragment layout via per-wave LDS tile
    {
      const int n = lane & 15, mh = (lane >> 4) << 3;
#pragma unroll
      for (int r = 0; r < 8; ++r) {
        ldsP[wv][mh + r][n]      = (__bf16)s[0][r];
        ldsP[wv][mh + r][16 + n] = (__bf16)s[1][r];
      }
    }
    __syncthreads();
    v16bf pa;
    {
      const int m = lane & 15, kk = (lane >> 4) << 3;
#pragma unroll
      for (int i = 0; i < 8; ++i) {
        pa[i]     = ldsP[wv][m][kk + i];
        pa[i + 8] = ldsP[wv][m][16 + kk + i];
      }
    }
    __syncthreads();

#pragma unroll
    for (int co = 0; co < 6; ++co) {
      v16bf vf = load_frag_b128(Vh, SEQ, 16 * co, kv0);
      o[co] = __builtin_amdgcn_wmma_f32_16x16x32_bf16(
          false, pa, false, vf, (short)0, o[co], false, false);
    }
  }

  float inv[8];
#pragma unroll
  for (int r = 0; r < 8; ++r) inv[r] = 1.0f / lstate[r];
#pragma unroll
  for (int co = 0; co < 6; ++co) {
    const int d = h * HDIM + 16 * co + (lane & 15);
#pragma unroll
    for (int r = 0; r < 8; ++r) {
      const int m = ((lane >> 4) << 3) + r;
      Ob[((size_t)b * SEQ + (q0 + m)) * EMB + d] = (__bf16)(o[co][r] * inv[r]);
    }
  }
}

// ---------------------------------------------------------------------------
// Launch: 6 converts -> 3 projection GEMMs -> flash attention -> output GEMM
// ---------------------------------------------------------------------------
extern "C" void kernel_launch(void* const* d_in, const int* in_sizes, int n_in,
                              void* d_out, int out_size, void* d_ws, size_t ws_size,
                              hipStream_t stream) {
  const float* xq = (const float*)d_in[0];
  const float* xkv = (const float*)d_in[1];
  const float* Wq = (const float*)d_in[2];
  const float* bq = (const float*)d_in[3];
  const float* Wk = (const float*)d_in[4];
  const float* bk = (const float*)d_in[5];
  const float* Wv = (const float*)d_in[6];
  const float* bv = (const float*)d_in[7];
  const float* Wo = (const float*)d_in[8];
  const float* bo = (const float*)d_in[9];

  char* ws = (char*)d_ws;
  size_t off = 0;
  auto carve = [&](size_t bytes) -> void* {
    void* p = ws + off;
    off += (bytes + 255) & ~(size_t)255;
    return p;
  };
  const size_t actB = (size_t)MROWS * EMB * sizeof(__bf16);
  const size_t wB   = (size_t)EMB * EMB * sizeof(__bf16);

  __bf16* xq_b  = (__bf16*)carve(actB);
  __bf16* xkv_b = (__bf16*)carve(actB);
  __bf16* wq_b  = (__bf16*)carve(wB);
  __bf16* wk_b  = (__bf16*)carve(wB);
  __bf16* wv_b  = (__bf16*)carve(wB);
  __bf16* wo_b  = (__bf16*)carve(wB);
  __bf16* Qbuf  = (__bf16*)carve(actB);   // [B,H,SEQ,96]
  __bf16* Kbuf  = (__bf16*)carve(actB);   // [B,H,SEQ,96]
  __bf16* Vtbuf = (__bf16*)carve(actB);   // [B,H,96,SEQ]
  __bf16* Obuf  = (__bf16*)carve(actB);   // [B,SEQ,EMB]

  const int nAct = MROWS * EMB, nW = EMB * EMB;
  cvt_f32_bf16<<<(nAct + 255) / 256, 256, 0, stream>>>(xq,  xq_b,  nAct);
  cvt_f32_bf16<<<(nAct + 255) / 256, 256, 0, stream>>>(xkv, xkv_b, nAct);
  cvt_f32_bf16<<<(nW + 255) / 256, 256, 0, stream>>>(Wq, wq_b, nW);
  cvt_f32_bf16<<<(nW + 255) / 256, 256, 0, stream>>>(Wk, wk_b, nW);
  cvt_f32_bf16<<<(nW + 255) / 256, 256, 0, stream>>>(Wv, wv_b, nW);
  cvt_f32_bf16<<<(nW + 255) / 256, 256, 0, stream>>>(Wo, wo_b, nW);

  dim3 gGemm(EMB / 64, MROWS / 256), blk(256);    // (12, 32)
  gemm_wmma<1><<<gGemm, blk, 0, stream>>>(xq_b,  wq_b, bq, (void*)Qbuf);
  gemm_wmma<1><<<gGemm, blk, 0, stream>>>(xkv_b, wk_b, bk, (void*)Kbuf);
  gemm_wmma<2><<<gGemm, blk, 0, stream>>>(xkv_b, wv_b, bv, (void*)Vtbuf);

  attn_wmma<<<dim3(SEQ / 128, BATCH * HEADS), blk, 0, stream>>>(Qbuf, Kbuf, Vtbuf, Obuf);

  gemm_wmma<0><<<gGemm, blk, 0, stream>>>(Obuf, wo_b, bo, d_out);
}
